// SimpleRGCN_84482006713255
// MI455X (gfx1250) — matrix operand order repaired and checked by reading, not compiled, not run on hardware
//
#include <hip/hip_runtime.h>
#include <hip/hip_bf16.h>
#include <math.h>

typedef __attribute__((ext_vector_type(16))) _Float16 v16h;
typedef __attribute__((ext_vector_type(8)))  float    v8f;

#define N_NODES 50000
#define N_RELS  16
#define EMB     128
#define N_BASES 16
#define N_EDGES 1000000

// ---------------------------------------------------------------------------
// Native CDNA5 float atomic-add (no return -> STOREcnt only). Guarantees
// GLOBAL_ATOMIC_ADD_F32 instead of a possible CAS-loop expansion of the
// generic atomic builtin. s_endpgm's implicit wait-idle covers completion.
// ---------------------------------------------------------------------------
__device__ __forceinline__ void global_atomic_fadd(float* addr, float val) {
  asm volatile("global_atomic_add_f32 %0, %1, off"
               :
               : "v"(addr), "v"(val)
               : "memory");
}

// ---------------------------------------------------------------------------
// WMMA helpers (wave32, v_wmma_f32_16x16x32_f16)
// ---------------------------------------------------------------------------

// A-matrix fragment (16 rows x 32 K) built from a row-major f32 row pointer.
// Contiguous k order used here (same as B): lane L (half = L>>4) holds, for row m = L&15:
//   idx 0..7  -> k = kbase + 16*half + idx
//   idx 8..15 -> k = kbase + 16*half + 8 + (idx-8)
// Both runs are 8 contiguous floats -> two pairs of float4 loads.
__device__ inline v16h load_a_frag(const float* __restrict__ row_ptr, int kbase, int half) {
  v16h a;
  const float4* p0 = (const float4*)(row_ptr + kbase + 16 * half);
  const float4* p1 = (const float4*)(row_ptr + kbase + 16 * half + 8);
#pragma unroll
  for (int q = 0; q < 2; ++q) {
    float4 f = p0[q];
    a[4 * q + 0] = (_Float16)f.x; a[4 * q + 1] = (_Float16)f.y;
    a[4 * q + 2] = (_Float16)f.z; a[4 * q + 3] = (_Float16)f.w;
  }
#pragma unroll
  for (int q = 0; q < 2; ++q) {
    float4 f = p1[q];
    a[8 + 4 * q + 0] = (_Float16)f.x; a[8 + 4 * q + 1] = (_Float16)f.y;
    a[8 + 4 * q + 2] = (_Float16)f.z; a[8 + 4 * q + 3] = (_Float16)f.w;
  }
  return a;
}

// B-matrix fragment (32 K x 16 cols) where B[k][n] = M[n][k] (i.e. B = M^T for
// a row-major M). Lane L owns column n = L&15; its 16 elements are
// k = kbase + 16*half + idx, a single contiguous 16-float run of row n of M.
__device__ inline v16h load_bT_frag(const float* __restrict__ row_ptr, int kstart) {
  v16h b;
  const float4* p = (const float4*)(row_ptr + kstart);
#pragma unroll
  for (int q = 0; q < 4; ++q) {
    float4 f = p[q];
    b[4 * q + 0] = (_Float16)f.x; b[4 * q + 1] = (_Float16)f.y;
    b[4 * q + 2] = (_Float16)f.z; b[4 * q + 3] = (_Float16)f.w;
  }
  return b;
}

__device__ inline v8f wmma_f16(v16h a, v16h b, v8f c) {
  // (neg_a, A, neg_b, B, c_mod, C, reuse_a, reuse_b)
  return __builtin_amdgcn_wmma_f32_16x16x32_f16(false, a, false, b, (short)0, c,
                                                false, false);
}

// ---------------------------------------------------------------------------
// Stage 1: per-node attention table.
//   kx = X K^T, qx = X Q^T (per 16-node tile, WMMA)
//   h  = kx (*) qx, staged in LDS in C/D layout
//   dots[m,r] = (h R^T)[m,r] / 128 ; vtab = exp(dots)    [N_NODES x 16]
// One wave per 16-node tile.
// ---------------------------------------------------------------------------
__global__ __launch_bounds__(32)
void rgcn_node_tables(const float* __restrict__ X,
                      const float* __restrict__ tokeys,
                      const float* __restrict__ toqueries,
                      const float* __restrict__ relations,
                      float* __restrict__ vtab) {
  __shared__ float hbuf[16 * EMB];
  const int lane = threadIdx.x;
  const int half = lane >> 4;
  const int m    = lane & 15;
  const int node_base = blockIdx.x * 16;

  // A fragments of the X tile for all 4 K-chunks (K = 128).
  v16h a[4];
  const float* xrow = X + (size_t)(node_base + m) * EMB;
#pragma unroll
  for (int kc = 0; kc < 4; ++kc) a[kc] = load_a_frag(xrow, kc * 32, half);

  const v8f vzero = {};
#pragma unroll
  for (int cb = 0; cb < 8; ++cb) {
    v8f ck = vzero, cq = vzero;
    const int col = cb * 16 + m;  // output column i == row of tokeys/toqueries
    const float* krow = tokeys    + (size_t)col * EMB;
    const float* qrow = toqueries + (size_t)col * EMB;
#pragma unroll
    for (int kc = 0; kc < 4; ++kc) {
      const int kstart = kc * 32 + 16 * half;
      ck = wmma_f16(a[kc], load_bT_frag(krow, kstart), ck);
      cq = wmma_f16(a[kc], load_bT_frag(qrow, kstart), cq);
    }
    // h tile in C/D layout: row = v + 8*half, col = cb*16 + (lane&15)
#pragma unroll
    for (int v = 0; v < 8; ++v)
      hbuf[(v + 8 * half) * EMB + cb * 16 + m] = ck[v] * cq[v];
  }
  __syncthreads();

  // dots = h @ relations^T / EMB ; one 16x16 output tile, K = 128.
  v8f d = vzero;
  const float* hrow = hbuf + m * EMB;                // A row for this lane
  const float* rrow = relations + (size_t)m * EMB;   // B col r == lane&15
#pragma unroll
  for (int kc = 0; kc < 4; ++kc) {
    v16h ha = load_a_frag(hrow, kc * 32, half);
    d = wmma_f16(ha, load_bT_frag(rrow, kc * 32 + 16 * half), d);
  }
#pragma unroll
  for (int v = 0; v < 8; ++v) {
    float val = __expf(d[v] * (1.0f / EMB));
    vtab[(size_t)(node_base + v + 8 * half) * N_RELS + m] = val;
  }
}

// ---------------------------------------------------------------------------
// Stage 2: segment row-sums. rowsum[rel*N + src] += vtab[src, rel]
// ---------------------------------------------------------------------------
__global__ __launch_bounds__(256)
void rgcn_rowsum(const int* __restrict__ src, const int* __restrict__ rel,
                 const float* __restrict__ vtab, float* __restrict__ rowsum) {
  int e = blockIdx.x * blockDim.x + threadIdx.x;
  if (e >= N_EDGES) return;
  int s = src[e], r = rel[e];
  float v = vtab[(size_t)s * N_RELS + r];
  global_atomic_fadd(&rowsum[(size_t)r * N_NODES + s], v);
}

// ---------------------------------------------------------------------------
// Stage 3: normalized message scatter into O[r, src, :] (128 lanes per edge).
// ---------------------------------------------------------------------------
__global__ __launch_bounds__(256)
void rgcn_scatter(const int* __restrict__ src, const int* __restrict__ rel,
                  const int* __restrict__ dst, const float* __restrict__ X,
                  const float* __restrict__ vtab, const float* __restrict__ rowsum,
                  float* __restrict__ O) {
  int e = blockIdx.x * 2 + (threadIdx.x >> 7);
  int j = threadIdx.x & 127;
  if (e >= N_EDGES) return;
  int s = src[e], r = rel[e], dn = dst[e];
  float coeff = vtab[(size_t)s * N_RELS + r] / rowsum[(size_t)r * N_NODES + s];
  float msg = coeff * X[(size_t)dn * EMB + j];
  global_atomic_fadd(&O[((size_t)r * N_NODES + s) * EMB + j], msg);
}

// ---------------------------------------------------------------------------
// Stage 4a: W[r,i,j] = sum_b comps[r,b]*bases_w[b,i,j], packed as f16 directly
// in per-lane WMMA B-fragment order for the big GEMM (k' = r*128 + j):
//   wtp[kc*4096 + cb*512 + lane*16 + idx],  k' = kc*32 + 16*(lane>>4) + idx,
//   col i = cb*16 + (lane&15).
// ---------------------------------------------------------------------------
__global__ __launch_bounds__(256)
void rgcn_pack_weights(const float* __restrict__ comps,
                       const float* __restrict__ bases_w,
                       _Float16* __restrict__ wtp) {
  int p = blockIdx.x * blockDim.x + threadIdx.x;
  if (p >= 2048 * 128) return;
  int idx  = p & 15;
  int lane = (p >> 4) & 31;
  int cb   = (p >> 9) & 7;
  int kc   = p >> 12;                      // 0..63
  int kk   = kc * 32 + 16 * (lane >> 4) + idx;  // k' in 0..2047
  int i    = cb * 16 + (lane & 15);
  int r    = kk >> 7;
  int j    = kk & 127;
  float w = 0.0f;
#pragma unroll
  for (int b = 0; b < N_BASES; ++b)
    w += comps[r * N_BASES + b] * bases_w[((size_t)b * EMB + i) * EMB + j];
  wtp[p] = (_Float16)w;
}

// ---------------------------------------------------------------------------
// Stage 4b: out[n,i] = relu( sum_{k'=0..2047} O'[n,k'] * W'[k',i] )
// where O'[n, r*128+j] = O[r,n,j]. A 50000 x 2048 x 128 GEMM; one wave per
// 16-node tile, full 16x128 accumulator tile in registers (8 x v8f).
// ---------------------------------------------------------------------------
__global__ __launch_bounds__(32)
void rgcn_transform(const float* __restrict__ O,
                    const _Float16* __restrict__ wtp,
                    float* __restrict__ out) {
  const int lane = threadIdx.x;
  const int half = lane >> 4;
  const int m    = lane & 15;
  const int node_base = blockIdx.x * 16;

  const v8f vzero = {};
  v8f acc[8];
#pragma unroll
  for (int cb = 0; cb < 8; ++cb) acc[cb] = vzero;

#pragma unroll 4
  for (int kc = 0; kc < 64; ++kc) {
    const int kbase = kc * 32;
    const int r  = kbase >> 7;    // relation for this K-chunk (32 divides 128)
    const int j0 = kbase & 127;   // column offset inside the O row
    const float* orow = O + ((size_t)r * N_NODES + node_base + m) * EMB;
    v16h a = load_a_frag(orow, j0, half);
    const v16h* bf = (const v16h*)wtp + kc * 256 + lane;  // pre-packed frags
#pragma unroll
    for (int cb = 0; cb < 8; ++cb)
      acc[cb] = wmma_f16(a, bf[cb * 32], acc[cb]);
  }

#pragma unroll
  for (int cb = 0; cb < 8; ++cb)
#pragma unroll
    for (int v = 0; v < 8; ++v) {
      float val = acc[cb][v];
      out[(size_t)(node_base + v + 8 * half) * EMB + cb * 16 + m] =
          val > 0.0f ? val : 0.0f;
    }
}

// ---------------------------------------------------------------------------
// Launch
// ---------------------------------------------------------------------------
extern "C" void kernel_launch(void* const* d_in, const int* in_sizes, int n_in,
                              void* d_out, int out_size, void* d_ws, size_t ws_size,
                              hipStream_t stream) {
  const int*   src       = (const int*)d_in[0];
  const int*   rel       = (const int*)d_in[1];
  const int*   dst       = (const int*)d_in[2];
  const float* X         = (const float*)d_in[3];
  const float* tokeys    = (const float*)d_in[4];
  const float* toqueries = (const float*)d_in[5];
  const float* relations = (const float*)d_in[6];
  const float* comps     = (const float*)d_in[7];
  const float* bases_w   = (const float*)d_in[8];
  float* out = (float*)d_out;

  // Workspace layout (all offsets 512B-aligned):
  //   O       : 16*50000*128 f32  = 409,600,000 B
  //   rowsum  : 16*50000     f32  =   3,200,000 B
  //   vtab    : 50000*16     f32  =   3,200,000 B
  //   wtp     : 2048*128     f16  =     524,288 B
  char* ws = (char*)d_ws;
  const size_t O_bytes  = (size_t)N_RELS * N_NODES * EMB * sizeof(float);
  const size_t RS_bytes = (size_t)N_RELS * N_NODES * sizeof(float);
  const size_t VT_bytes = (size_t)N_NODES * N_RELS * sizeof(float);
  float*    O      = (float*)ws;
  float*    rowsum = (float*)(ws + O_bytes);
  float*    vtab   = (float*)(ws + O_bytes + RS_bytes);
  _Float16* wtp    = (_Float16*)(ws + O_bytes + RS_bytes + VT_bytes);

  // O and rowsum are contiguous: zero both with one capturable memset.
  hipMemsetAsync(O, 0, O_bytes + RS_bytes, stream);

  rgcn_node_tables<<<N_NODES / 16, 32, 0, stream>>>(X, tokeys, toqueries,
                                                    relations, vtab);
  rgcn_pack_weights<<<(2048 * 128) / 256, 256, 0, stream>>>(comps, bases_w, wtp);
  rgcn_rowsum<<<(N_EDGES + 255) / 256, 256, 0, stream>>>(src, rel, vtab, rowsum);
  rgcn_scatter<<<N_EDGES / 2, 256, 0, stream>>>(src, rel, dst, X, vtab, rowsum, O);
  rgcn_transform<<<N_NODES / 16, 32, 0, stream>>>(O, wtp, out);
}
